// Weird_Attention_4690104287848
// MI455X (gfx1250) — compile-verified, matrix operand order checked by reading, not run on hardware
//
#include <hip/hip_runtime.h>
#include <cstdint>

// ---------------------------------------------------------------------------
// Weird_Attention for MI455X (gfx1250, wave32, WMMA f16 -> f32 accumulate)
//
// Algebraic rewrite:  (probs_g @ probs_p) @ vp == probs_g @ (probs_p @ vp)
//   => last = probs_g @ (vg + attn(qp,kp,vp));  two flash-attention passes,
//      no O(S^3) bridge stage, no SxS probability buffers in memory.
// ---------------------------------------------------------------------------

#define B_SZ 4
#define S_LEN 1024
#define D_MOD 1024
#define H_NUM 16
#define DH 64
#define DFF 4096
#define M_ROWS (B_SZ * S_LEN)   // 4096

typedef __attribute__((ext_vector_type(16))) _Float16 v16h;
typedef __attribute__((ext_vector_type(8)))  _Float16 v8h;
typedef __attribute__((ext_vector_type(8)))  float    v8f;
typedef __attribute__((ext_vector_type(4)))  int      v4i;

__device__ __forceinline__ v16h mk16(v8h lo, v8h hi) {
  v16h r;
#pragma unroll
  for (int i = 0; i < 8; ++i) { r[i] = lo[i]; r[i + 8] = hi[i]; }
  return r;
}
__device__ __forceinline__ v8f zero8() {
  v8f z;
#pragma unroll
  for (int i = 0; i < 8; ++i) z[i] = 0.0f;
  return z;
}
__device__ __forceinline__ v8f wmma_f16(v16h a, v16h b, v8f c) {
  return __builtin_amdgcn_wmma_f32_16x16x32_f16(false, a, false, b,
                                                (short)0, c, false, false);
}

// ---- CDNA5 async global->LDS copy (ASYNCcnt path), with safe fallback ----
// clang prints addrspace(1) as "__device__" in HIP mode; the builtin wants
// int4 pointers qualified with AS1 (global src) / AS3 (LDS dst).
#if __has_builtin(__builtin_amdgcn_global_load_async_to_lds_b128)
#define HAVE_ASYNC_LDS 1
typedef __attribute__((address_space(1))) v4i g_v4i;
typedef __attribute__((address_space(3))) v4i l_v4i;
__device__ __forceinline__ void async_ld_b128(const void* g, void* l) {
  __builtin_amdgcn_global_load_async_to_lds_b128(
      (g_v4i*)(uintptr_t)g, (l_v4i*)(uint32_t)(uintptr_t)l, 0, 0);
}
#else
#define HAVE_ASYNC_LDS 0
#endif

__device__ __forceinline__ void wait_async0() {
#if __has_builtin(__builtin_amdgcn_s_wait_asynccnt)
  __builtin_amdgcn_s_wait_asynccnt(0);
#else
  asm volatile("s_wait_asynccnt 0" ::: "memory");
#endif
}

// ---------------------------------------------------------------------------
// GEMM:  C[M,N] = act( A[M,K] @ W[N,K]^T + bias )   ("NT": both K-contiguous)
// A: f32 or f16 row-major; W: f32 row-major [N,K]; C: f16 or f32.
// Block tile 128(M) x 64(N), BK=64; 8 waves, each wave a 16x64 strip
// => 8 WMMAs per wave between each pair of barriers.
// ---------------------------------------------------------------------------
template <bool AF32, bool GELU, bool OUTF32>
__global__ __launch_bounds__(256) void gemm_nt_kernel(
    const void* __restrict__ Ap, const float* __restrict__ W,
    const float* __restrict__ bias, void* __restrict__ Cp,
    int M, int N, int K) {
  __shared__ _Float16 As[128][72];  // 144B row stride (16B aligned)
  __shared__ _Float16 Bs[64][72];

  const int tid = threadIdx.x;
  const int wv = tid >> 5, lane = tid & 31;
  const int part = lane >> 4, r16 = lane & 15;
  const int bm = blockIdx.x * 128, bn = blockIdx.y * 64;

  v8f acc[4];
#pragma unroll
  for (int nt = 0; nt < 4; ++nt) acc[nt] = zero8();

  for (int k0 = 0; k0 < K; k0 += 64) {
    // ---- stage A tile [128 x 64] ----
    if (AF32) {
      const float* A = (const float*)Ap;
#pragma unroll
      for (int i = 0; i < 8; ++i) {
        int c = tid * 8 + i;
        int row = c >> 4, col = (c & 15) * 4;
        const float4 f = *(const float4*)(A + (size_t)(bm + row) * K + k0 + col);
        As[row][col + 0] = (_Float16)f.x;
        As[row][col + 1] = (_Float16)f.y;
        As[row][col + 2] = (_Float16)f.z;
        As[row][col + 3] = (_Float16)f.w;
      }
    } else {
      const _Float16* A = (const _Float16*)Ap;
#pragma unroll
      for (int i = 0; i < 4; ++i) {
        int c = tid * 4 + i;
        int row = c >> 3, col = (c & 7) * 8;
        const _Float16* src = A + (size_t)(bm + row) * K + k0 + col;
#if HAVE_ASYNC_LDS
        async_ld_b128(src, &As[row][col]);      // ASYNCcnt path, no VGPRs
#else
        *(v8h*)&As[row][col] = *(const v8h*)src;
#endif
        if (i == 0 && k0 + 64 < K) __builtin_prefetch(src + 64, 0, 1);
      }
    }
    // ---- stage W tile [64 x 64] (fp32 -> f16 converts in VALU) ----
#pragma unroll
    for (int i = 0; i < 4; ++i) {
      int c = tid * 4 + i;
      int row = c >> 4, col = (c & 15) * 4;
      const float* src = W + (size_t)(bn + row) * K + k0 + col;
      const float4 f = *(const float4*)src;
      Bs[row][col + 0] = (_Float16)f.x;
      Bs[row][col + 1] = (_Float16)f.y;
      Bs[row][col + 2] = (_Float16)f.z;
      Bs[row][col + 3] = (_Float16)f.w;
      if (i == 0 && k0 + 64 < K) __builtin_prefetch(src + 64, 0, 1);
    }
#if HAVE_ASYNC_LDS
    if (!AF32) wait_async0();
#endif
    __syncthreads();

    // ---- fragments + 8 WMMAs ----
    const int arow = wv * 16 + r16;
    v16h af[2];
#pragma unroll
    for (int c = 0; c < 2; ++c)
      af[c] = mk16(*(const v8h*)&As[arow][c * 32 + part * 8],
                   *(const v8h*)&As[arow][c * 32 + part * 8 + 16]);
#pragma unroll
    for (int nt = 0; nt < 4; ++nt) {
      const int brow = nt * 16 + r16;
      const v16h b0 = mk16(*(const v8h*)&Bs[brow][part * 8],
                           *(const v8h*)&Bs[brow][part * 8 + 16]);
      const v16h b1 = mk16(*(const v8h*)&Bs[brow][32 + part * 8],
                           *(const v8h*)&Bs[brow][32 + part * 8 + 16]);
      acc[nt] = wmma_f16(af[0], b0, acc[nt]);
      acc[nt] = wmma_f16(af[1], b1, acc[nt]);
    }
    __syncthreads();
  }

  // ---- epilogue: bias (+GELU), store ----
#pragma unroll
  for (int nt = 0; nt < 4; ++nt) {
    const int n = bn + nt * 16 + r16;
    const float bv = bias[n];
#pragma unroll
    for (int r = 0; r < 8; ++r) {
      const int m = bm + wv * 16 + r + 8 * part;
      float v = acc[nt][r] + bv;
      if (GELU) v = 0.5f * v * (1.0f + erff(v * 0.70710678118654752f));
      if (OUTF32)
        ((float*)Cp)[(size_t)m * N + n] = v;
      else
        ((_Float16*)Cp)[(size_t)m * N + n] = (_Float16)v;
    }
  }
}

// ---------------------------------------------------------------------------
// Flash attention per (b, h): Out = softmax(Q K^T / 8) @ V  [+ addend]
// Q/K/V/Out/addend are f16 in merged [B*S, D] layout (row stride D, head
// column offset h*64). Block = 8 waves, each wave owns 16 query rows.
// K-chunks of 64: 8 WMMA (QK^T) + 8 WMMA (P.V) per chunk per wave.
// ---------------------------------------------------------------------------
__global__ __launch_bounds__(256) void attn_kernel(
    const _Float16* __restrict__ Q, const _Float16* __restrict__ Kp,
    const _Float16* __restrict__ V, const _Float16* __restrict__ addend,
    _Float16* __restrict__ Out) {
  __shared__ _Float16 Vt[64][72];       // V^T chunk: [dh][k], 144B row stride
  __shared__ _Float16 Pl[8][16][72];    // per-wave P tile: [m][k]

  const int tid = threadIdx.x;
  const int wv = tid >> 5, lane = tid & 31;
  const int part = lane >> 4, r16 = lane & 15;
  const int b = blockIdx.z, h = blockIdx.y, qb = blockIdx.x;
  const size_t base = ((size_t)b * S_LEN) * D_MOD + (size_t)h * DH;
  const int q0 = qb * 128 + wv * 16;

  // Q fragments (resident in registers for the whole kernel)
  v16h qf[2];
  {
    const _Float16* qp = Q + base + (size_t)(q0 + r16) * D_MOD;
#pragma unroll
    for (int c = 0; c < 2; ++c)
      qf[c] = mk16(*(const v8h*)(qp + c * 32 + part * 8),
                   *(const v8h*)(qp + c * 32 + part * 8 + 16));
  }

  float rm[8], rl[8];
  v8f o[4];
#pragma unroll
  for (int r = 0; r < 8; ++r) { rm[r] = -1e30f; rl[r] = 0.0f; }
#pragma unroll
  for (int nt = 0; nt < 4; ++nt) o[nt] = zero8();

  for (int kc = 0; kc < S_LEN; kc += 64) {
    __syncthreads();  // protect Vt/Pl from previous iteration's readers
    // ---- cooperative transpose-stage of V chunk ----
#pragma unroll
    for (int i = 0; i < 2; ++i) {
      int c = tid * 2 + i;
      int kk = c >> 3, d0 = (c & 7) * 8;
      const _Float16* src = V + base + (size_t)(kc + kk) * D_MOD + d0;
      const v8h v = *(const v8h*)src;
      if (i == 0 && kc + 64 < S_LEN)
        __builtin_prefetch(src + (size_t)64 * D_MOD, 0, 1);
#pragma unroll
      for (int j = 0; j < 8; ++j) Vt[d0 + j][kk] = v[j];
    }
    __syncthreads();

    // ---- scores S = Q K^T (K fragments straight from global/L2) ----
    v8f s[4];
#pragma unroll
    for (int nt = 0; nt < 4; ++nt) {
      const _Float16* kp =
          Kp + base + (size_t)(kc + nt * 16 + r16) * D_MOD;
      const v16h k0f = mk16(*(const v8h*)(kp + part * 8),
                            *(const v8h*)(kp + part * 8 + 16));
      const v16h k1f = mk16(*(const v8h*)(kp + 32 + part * 8),
                            *(const v8h*)(kp + 32 + part * 8 + 16));
      v8f z = zero8();
      z = wmma_f16(qf[0], k0f, z);
      s[nt] = wmma_f16(qf[1], k1f, z);
    }
#pragma unroll
    for (int nt = 0; nt < 4; ++nt)
#pragma unroll
      for (int r = 0; r < 8; ++r) s[nt][r] *= 0.125f;  // 1/sqrt(64)

    // ---- online softmax (rows spread over 16 lanes of a half-wave) ----
    float mnew[8];
#pragma unroll
    for (int r = 0; r < 8; ++r) {
      float m = s[0][r];
#pragma unroll
      for (int nt = 1; nt < 4; ++nt) m = fmaxf(m, s[nt][r]);
      mnew[r] = m;
    }
#pragma unroll
    for (int msk = 8; msk >= 1; msk >>= 1)
#pragma unroll
      for (int r = 0; r < 8; ++r)
        mnew[r] = fmaxf(mnew[r], __shfl_xor(mnew[r], msk));
#pragma unroll
    for (int r = 0; r < 8; ++r) {
      const float mo = rm[r];
      const float mN = fmaxf(mo, mnew[r]);
      rm[r] = mN;
      const float f = __expf(mo - mN);
      rl[r] *= f;
#pragma unroll
      for (int nt = 0; nt < 4; ++nt) o[nt][r] *= f;
    }
    float psum[8];
#pragma unroll
    for (int r = 0; r < 8; ++r) psum[r] = 0.0f;
#pragma unroll
    for (int nt = 0; nt < 4; ++nt)
#pragma unroll
      for (int r = 0; r < 8; ++r) {
        const float p = __expf(s[nt][r] - rm[r]);
        s[nt][r] = p;
        psum[r] += p;
      }
#pragma unroll
    for (int msk = 8; msk >= 1; msk >>= 1)
#pragma unroll
      for (int r = 0; r < 8; ++r) psum[r] += __shfl_xor(psum[r], msk);
#pragma unroll
    for (int r = 0; r < 8; ++r) rl[r] += psum[r];

    // ---- write P (C-layout) to LDS, re-read as A-layout fragments ----
#pragma unroll
    for (int nt = 0; nt < 4; ++nt)
#pragma unroll
      for (int r = 0; r < 8; ++r)
        Pl[wv][r + 8 * part][nt * 16 + r16] = (_Float16)s[nt][r];
    __syncthreads();

    v16h pf[2];
    {
      const _Float16* pp = &Pl[wv][r16][0];
#pragma unroll
      for (int c = 0; c < 2; ++c)
        pf[c] = mk16(*(const v8h*)(pp + c * 32 + part * 8),
                     *(const v8h*)(pp + c * 32 + part * 8 + 16));
    }
    // ---- O += P @ V  (B fragments from transposed Vt) ----
#pragma unroll
    for (int nt = 0; nt < 4; ++nt) {
      const _Float16* vp = &Vt[nt * 16 + r16][0];
      const v16h v0 = mk16(*(const v8h*)(vp + part * 8),
                           *(const v8h*)(vp + part * 8 + 16));
      const v16h v1 = mk16(*(const v8h*)(vp + 32 + part * 8),
                           *(const v8h*)(vp + 32 + part * 8 + 16));
      o[nt] = wmma_f16(pf[0], v0, o[nt]);
      o[nt] = wmma_f16(pf[1], v1, o[nt]);
    }
  }

  // ---- finalize: O / l  (+ optional fused addend), f16 store ----
  float inv[8];
#pragma unroll
  for (int r = 0; r < 8; ++r) inv[r] = 1.0f / rl[r];
#pragma unroll
  for (int nt = 0; nt < 4; ++nt)
#pragma unroll
    for (int r = 0; r < 8; ++r) {
      const size_t idx =
          base + (size_t)(q0 + r + 8 * part) * D_MOD + nt * 16 + r16;
      float v = o[nt][r] * inv[r];
      if (addend) v += (float)addend[idx];
      Out[idx] = (_Float16)v;
    }
}

// ---------------------------------------------------------------------------
// Orchestration.  Workspace layout: 9 f16 buffers of 4M halves (72 MB total).
//  0: glob (later ctx)   1: local (later dense_out)   2: dup (later fc2_out)
//  3: qg   4: kg   5: vg (in-place -> vg + attn_local = v2)
//  6: qp   7: kp   8: vp        fc1_out (16M halves) reuses bufs 3..6.
// ---------------------------------------------------------------------------
extern "C" void kernel_launch(void* const* d_in, const int* in_sizes, int n_in,
                              void* d_out, int out_size, void* d_ws,
                              size_t ws_size, hipStream_t stream) {
  (void)in_sizes; (void)n_in; (void)out_size; (void)ws_size;

  const float* global_feat = (const float*)d_in[0];
  const float* local_feat  = (const float*)d_in[1];
  const float* text_feat   = (const float*)d_in[2];
  const float* ge_w = (const float*)d_in[3];  const float* ge_b = (const float*)d_in[4];
  const float* le1_w = (const float*)d_in[5]; const float* le1_b = (const float*)d_in[6];
  const float* le2_w = (const float*)d_in[7]; const float* le2_b = (const float*)d_in[8];
  const float* qg_w = (const float*)d_in[9];  const float* qg_b = (const float*)d_in[10];
  const float* kg_w = (const float*)d_in[11]; const float* kg_b = (const float*)d_in[12];
  const float* vg_w = (const float*)d_in[13]; const float* vg_b = (const float*)d_in[14];
  const float* qp_w = (const float*)d_in[15]; const float* qp_b = (const float*)d_in[16];
  const float* kp_w = (const float*)d_in[17]; const float* kp_b = (const float*)d_in[18];
  const float* vp_w = (const float*)d_in[19]; const float* vp_b = (const float*)d_in[20];
  const float* dense_w = (const float*)d_in[21]; const float* dense_b = (const float*)d_in[22];
  const float* ml_w = (const float*)d_in[23]; const float* ml_b = (const float*)d_in[24];
  const float* fc1_w = (const float*)d_in[25]; const float* fc1_b = (const float*)d_in[26];
  const float* fc2_w = (const float*)d_in[27]; const float* fc2_b = (const float*)d_in[28];

  const size_t M4 = (size_t)M_ROWS * D_MOD;  // 4M halves
  auto buf = [&](int i) { return (_Float16*)d_ws + (size_t)i * M4; };

  const dim3 blk(256);
  const dim3 gD(M_ROWS / 128, D_MOD / 64);  // N = 1024
  const dim3 gF(M_ROWS / 128, DFF / 64);    // N = 4096
  const dim3 gA(S_LEN / 128, H_NUM, B_SZ);  // attention

  // embeddings (fp32 activations in, f16 out)
  gemm_nt_kernel<true , false, false><<<gD, blk, 0, stream>>>(global_feat, ge_w,  ge_b,  buf(0), M_ROWS, D_MOD, D_MOD);
  gemm_nt_kernel<true , false, false><<<gD, blk, 0, stream>>>(local_feat,  le2_w, le2_b, buf(1), M_ROWS, D_MOD, D_MOD);
  gemm_nt_kernel<true , false, false><<<gD, blk, 0, stream>>>(local_feat,  le1_w, le1_b, buf(2), M_ROWS, D_MOD, D_MOD);
  // Q/K/V projections
  gemm_nt_kernel<false, false, false><<<gD, blk, 0, stream>>>(buf(0), qg_w, qg_b, buf(3), M_ROWS, D_MOD, D_MOD);
  gemm_nt_kernel<false, false, false><<<gD, blk, 0, stream>>>(buf(1), kg_w, kg_b, buf(4), M_ROWS, D_MOD, D_MOD);
  gemm_nt_kernel<false, false, false><<<gD, blk, 0, stream>>>(buf(1), vg_w, vg_b, buf(5), M_ROWS, D_MOD, D_MOD);
  gemm_nt_kernel<false, false, false><<<gD, blk, 0, stream>>>(buf(2), qp_w, qp_b, buf(6), M_ROWS, D_MOD, D_MOD);
  gemm_nt_kernel<true , false, false><<<gD, blk, 0, stream>>>(text_feat, kp_w, kp_b, buf(7), M_ROWS, D_MOD, D_MOD);
  gemm_nt_kernel<true , false, false><<<gD, blk, 0, stream>>>(text_feat, vp_w, vp_b, buf(8), M_ROWS, D_MOD, D_MOD);
  // attn1: v2 = vg + softmax(qp kp^T) vp   (in-place on vg buffer)
  attn_kernel<<<gA, blk, 0, stream>>>(buf(6), buf(7), buf(8), buf(5), buf(5));
  // attn2: ctx = softmax(qg kg^T) v2       (== glb_ctx + plb_ctx, merged)
  attn_kernel<<<gA, blk, 0, stream>>>(buf(3), buf(4), buf(5), nullptr, buf(0));
  // dense -> MLP(GELU) -> ml
  gemm_nt_kernel<false, false, false><<<gD, blk, 0, stream>>>(buf(0), dense_w, dense_b, buf(1), M_ROWS, D_MOD, D_MOD);
  gemm_nt_kernel<false, true , false><<<gF, blk, 0, stream>>>(buf(1), fc1_w, fc1_b, buf(3), M_ROWS, DFF, D_MOD);
  gemm_nt_kernel<false, false, false><<<gD, blk, 0, stream>>>(buf(3), fc2_w, fc2_b, buf(2), M_ROWS, D_MOD, DFF);
  gemm_nt_kernel<false, false, true ><<<gD, blk, 0, stream>>>(buf(2), ml_w, ml_b, (float*)d_out, M_ROWS, D_MOD, D_MOD);
}
